// crossAttention_88115549044809
// MI455X (gfx1250) — matrix operand order repaired and checked by reading, not compiled, not run on hardware
//
#include <hip/hip_runtime.h>

// ---------------------------------------------------------------------------
// CDNA5 (gfx1250) XCiT-style cross attention.
// Heavy math on v_wmma_f32_16x16x32_f16 (wave32 WMMA), f32->f16 operands,
// f32 accumulation.  Fragment layouts per CDNA5 ISA 7.12.2:
//   A (16x32 f16): lane<16 -> row M=lane&15, elems K = hi*16+{0..15} (this code's order, not ISA table)
//   B (32x16 f16): lane&15 = N column, element t -> K = (lane>>4)*16 + t (same order as A, code's choice)
//   C/D (16x16 f32): N = lane&15, M = vgpr + 8*(lane>>4)
// B tiles are stored TRANSPOSED in LDS so every lane's 16 fragment elements
// are contiguous -> 2x ds_load_b128 per fragment (instead of 16x ds_load_u16).
// ---------------------------------------------------------------------------

typedef __attribute__((ext_vector_type(16))) _Float16 v16h;
typedef __attribute__((ext_vector_type(8)))  _Float16 v8h;
typedef __attribute__((ext_vector_type(8)))  float    v8f;

#define HW    16384
#define IMGH  128
#define IMGW  128
#define CDIM  384
#define NB    4
#define NHEAD 8
#define HD    48   // head dim = 384/8

static __device__ __forceinline__ v8f wmma_f16(const v16h& a, const v16h& b, const v8f& c) {
  return __builtin_amdgcn_wmma_f32_16x16x32_f16(false, a, false, b, (short)0, c, false, false);
}
static __device__ __forceinline__ v16h concat8(v8h lo, v8h hi) {
  return __builtin_shufflevector(lo, hi, 0,1,2,3,4,5,6,7,8,9,10,11,12,13,14,15);
}
// two 16B vector loads + concat (lowers to 2x b128 loads)
static __device__ __forceinline__ v16h load_frag(const _Float16* p0, const _Float16* p1) {
  return concat8(*(const v8h*)p0, *(const v8h*)p1);
}

// ---------------------------------------------------------------------------
// GEMM  Y[b] = W(MxK) * X[b](KxN),  f32 in/out, f16 WMMA inside.
// Block tile 64(M) x 128(N), K-step 32, 256 threads = 8 waves (4x2).
// Software pipelined: next k-tile prefetched from global into registers while
// the current tile's WMMAs run; B tile transposed in LDS (stride 40 halfs,
// 80B = 16B aligned rows) for vector fragment loads.
// ---------------------------------------------------------------------------
__global__ __launch_bounds__(256)
void gemm_wx_f32(const float* __restrict__ W, const float* __restrict__ X,
                 float* __restrict__ Y, int M, int K, int N,
                 long strideXb, long strideYb)
{
  __shared__ _Float16 As [64][32];        // row-major A tile
  __shared__ _Float16 BsT[128][40];       // TRANSPOSED B tile [col][k]

  const int b = blockIdx.z;
  const float* Xb = X + (long)b * strideXb;
  float*       Yb = Y + (long)b * strideYb;

  const int rowBlock = blockIdx.y * 64;
  const int colBlock = blockIdx.x * 128;
  const int tid  = threadIdx.x;
  const int lane = tid & 31;
  const int wv   = tid >> 5;       // 0..7
  const int wm   = wv & 3;         // wave M-tile
  const int wn   = wv >> 2;        // wave N-half
  const int lr   = lane & 15;
  const int hi   = lane >> 4;

  const int aRow = tid >> 2;            // 0..63
  const int aCol = (tid & 3) * 8;       // 0,8,16,24
  const int bRow = tid >> 3;            // 0..31  (k index)
  const int bCol = (tid & 7) * 16;      // 0..112 (n index)

  const float* aSrc = W  + (long)(rowBlock + aRow) * K + aCol;
  const float* bSrc = Xb + (long)bRow * N + colBlock + bCol;

  v8f acc[4] = {};
  const int nK = K / 32;

  float4 pa[2], pb[4];
  // prologue: fetch k-tile 0
  pa[0] = ((const float4*)aSrc)[0];
  pa[1] = ((const float4*)aSrc)[1];
  #pragma unroll
  for (int i = 0; i < 4; ++i) pb[i] = ((const float4*)bSrc)[i];

  for (int ks = 0; ks < nK; ++ks) {
    // stage prefetched tile into LDS (f32 -> f16)
    {
      v8h av;
      const float* pf = (const float*)pa;
      #pragma unroll
      for (int i = 0; i < 8; ++i) av[i] = (_Float16)pf[i];
      *(v8h*)&As[aRow][aCol] = av;                       // ds_store_b128
      const float* bf = (const float*)pb;
      #pragma unroll
      for (int i = 0; i < 16; ++i) BsT[bCol + i][bRow] = (_Float16)bf[i];
    }
    __syncthreads();

    // prefetch next k-tile while WMMAs execute
    if (ks + 1 < nK) {
      const float* aN = aSrc + (ks + 1) * 32;
      const float* bN = bSrc + (long)(ks + 1) * 32 * N;
      pa[0] = ((const float4*)aN)[0];
      pa[1] = ((const float4*)aN)[1];
      #pragma unroll
      for (int i = 0; i < 4; ++i) pb[i] = ((const float4*)bN)[i];
    }

    const _Float16* ap = &As[wm * 16 + lr][hi * 16];
    const v16h a = load_frag(ap, ap + 8);
    #pragma unroll
    for (int j = 0; j < 4; ++j) {
      const _Float16* bp = &BsT[wn * 64 + j * 16 + lr][hi * 16];
      const v16h bfr = load_frag(bp, bp + 8);
      acc[j] = wmma_f16(a, bfr, acc[j]);
    }
    __syncthreads();
  }

  #pragma unroll
  for (int j = 0; j < 4; ++j) {
    const int gr = rowBlock + wm * 16 + hi * 8;
    const int gc = colBlock + wn * 64 + j * 16 + lr;
    #pragma unroll
    for (int r = 0; r < 8; ++r)
      Yb[(long)(gr + r) * N + gc] = acc[j][r];
  }
}

// ---------------------------------------------------------------------------
// Depthwise 3x3, SAME padding (memory bound VALU kernel).
// ---------------------------------------------------------------------------
__global__ __launch_bounds__(256)
void dwconv3x3(const float* __restrict__ in, float* __restrict__ out,
               const float* __restrict__ wdw, int chOffset, int inC, int outC)
{
  long idx = (long)blockIdx.x * blockDim.x + threadIdx.x;
  const long total = (long)NB * outC * HW;
  if (idx >= total) return;
  const int x = idx & (IMGW - 1);
  long t = idx >> 7;
  const int y = t & (IMGH - 1);
  t >>= 7;
  const int c = (int)(t % outC);
  const int b = (int)(t / outC);

  const float* w   = wdw + (long)(chOffset + c) * 9;
  const float* src = in + ((long)b * inC + chOffset + c) * (long)HW;

  float s = 0.f;
  #pragma unroll
  for (int ky = -1; ky <= 1; ++ky) {
    const int yy = y + ky;
    if (yy < 0 || yy >= IMGH) continue;
    #pragma unroll
    for (int kx = -1; kx <= 1; ++kx) {
      const int xx = x + kx;
      if (xx < 0 || xx >= IMGW) continue;
      s += w[(ky + 1) * 3 + (kx + 1)] * src[(long)yy * IMGW + xx];
    }
  }
  out[idx] = s;
}

// ---------------------------------------------------------------------------
// Per-row L2 normalize over HW and/or convert f32 -> f16.
// ---------------------------------------------------------------------------
__global__ __launch_bounds__(256)
void row_to_half(const float* __restrict__ in, _Float16* __restrict__ out,
                 int rowsPerBatchIn, int rowOffset, int rowsOut, int doNorm)
{
  const int r = blockIdx.x;
  const int b = blockIdx.y;
  const float* src = in + ((long)b * rowsPerBatchIn + rowOffset + r) * (long)HW;
  _Float16*    dst = out + ((long)b * rowsOut + r) * (long)HW;

  __shared__ float red[256];
  float scale = 1.f;
  if (doNorm) {
    float s = 0.f;
    for (int i = threadIdx.x; i < HW; i += 256) { const float v = src[i]; s += v * v; }
    red[threadIdx.x] = s;
    __syncthreads();
    for (int o = 128; o > 0; o >>= 1) {
      if (threadIdx.x < o) red[threadIdx.x] += red[threadIdx.x + o];
      __syncthreads();
    }
    scale = 1.f / fmaxf(sqrtf(red[0]), 1e-12f);
  }
  for (int i = threadIdx.x; i < HW; i += 256)
    dst[i] = (_Float16)(src[i] * scale);
}

// ---------------------------------------------------------------------------
// attn = softmax( (Qm . Kn^T) * temperature ) per (b, head).
// M=N=48, K=HW: K split across 8 waves, 3x3 WMMA tiles per wave, LDS atomic
// f32 reduction, fused softmax, f16 output padded to 48x64.
// Fragments load straight from global (contiguous, 2x b128 per fragment).
// ---------------------------------------------------------------------------
__global__ __launch_bounds__(256)
void attn_qk_softmax(const _Float16* __restrict__ Qm, const _Float16* __restrict__ Kn,
                     const float* __restrict__ temperature, _Float16* __restrict__ attnH)
{
  const int h = blockIdx.x;
  const int b = blockIdx.y;
  const int lane = threadIdx.x & 31;
  const int wv   = threadIdx.x >> 5;
  const int lr   = lane & 15;
  const int hi   = lane >> 4;

  const long base = ((long)b * CDIM + h * HD) * (long)HW;
  const _Float16* Q = Qm + base;
  const _Float16* K = Kn + base;

  v8f acc[3][3] = {};
  const int kBeg = wv * 2048, kEnd = kBeg + 2048;
  for (int k0 = kBeg; k0 < kEnd; k0 += 32) {
    v16h a[3], bf[3];
    #pragma unroll
    for (int m = 0; m < 3; ++m) {
      const _Float16* src = Q + (long)(m * 16 + lr) * HW + k0 + hi * 16;
      a[m] = load_frag(src, src + 8);
    }
    #pragma unroll
    for (int n = 0; n < 3; ++n) {
      const _Float16* src = K + (long)(n * 16 + lr) * HW + k0 + hi * 16;
      bf[n] = load_frag(src, src + 8);
    }
    #pragma unroll
    for (int m = 0; m < 3; ++m)
      #pragma unroll
      for (int n = 0; n < 3; ++n)
        acc[m][n] = wmma_f16(a[m], bf[n], acc[m][n]);
  }

  __shared__ float S[HD][HD + 1];
  for (int i = threadIdx.x; i < HD * (HD + 1); i += 256) (&S[0][0])[i] = 0.f;
  __syncthreads();
  #pragma unroll
  for (int m = 0; m < 3; ++m)
    #pragma unroll
    for (int n = 0; n < 3; ++n) {
      const int row = m * 16 + hi * 8;
      const int col = n * 16 + lr;
      #pragma unroll
      for (int r = 0; r < 8; ++r)
        atomicAdd(&S[row + r][col], acc[m][n][r]);
    }
  __syncthreads();

  if (threadIdx.x < HD) {
    const int r = threadIdx.x;
    const float temp = temperature[h];
    float mx = -3.4e38f;
    for (int c = 0; c < HD; ++c) { const float v = S[r][c] * temp; mx = fmaxf(mx, v); }
    float sum = 0.f;
    for (int c = 0; c < HD; ++c) sum += __expf(S[r][c] * temp - mx);
    const float inv = 1.f / sum;
    _Float16* dst = attnH + (((long)b * NHEAD + h) * HD + r) * 64;
    for (int c = 0; c < HD; ++c) dst[c] = (_Float16)(__expf(S[r][c] * temp - mx) * inv);
    for (int c = HD; c < 64; ++c) dst[c] = (_Float16)0.f;
  }
}

// ---------------------------------------------------------------------------
// out = attn(48x48 padded to 48x64) . V(48xHW) per (b, head).
// V tile TRANSPOSED in LDS [col][k] (rows 48..63 zero), vector fragment loads.
// ---------------------------------------------------------------------------
__global__ __launch_bounds__(256)
void attn_v(const _Float16* __restrict__ attnH, const _Float16* __restrict__ V,
            float* __restrict__ out)
{
  const int nb = blockIdx.x;          // 128-column block
  const int h  = blockIdx.y;
  const int b  = blockIdx.z;
  const int tid  = threadIdx.x;
  const int lane = tid & 31, wv = tid >> 5;
  const int lr = lane & 15, hi = lane >> 4;
  const int col0 = nb * 128;
  const long vbase = ((long)b * CDIM + h * HD) * (long)HW;

  __shared__ _Float16 VsT[128][64];   // [col][k], k rows 48..63 zeroed

  // coalesced global reads (v8h) + transposed scalar LDS stores
  #pragma unroll
  for (int chunk = 0; chunk < 4; ++chunk) {
    const int idx = chunk * 2048 + tid * 8;        // 8 consecutive cols
    const int r = idx >> 7;                        // k row 0..63
    const int c = idx & 127;
    if (r < HD) {
      const v8h vv = *(const v8h*)(V + vbase + (long)r * HW + col0 + c);
      #pragma unroll
      for (int i = 0; i < 8; ++i) VsT[c + i][r] = vv[i];
    } else {
      #pragma unroll
      for (int i = 0; i < 8; ++i) VsT[c + i][r] = (_Float16)0.f;
    }
  }
  __syncthreads();

  const _Float16* A = attnH + ((long)b * NHEAD + h) * HD * 64;
  v8f acc[3] = {};
  #pragma unroll
  for (int ks = 0; ks < 2; ++ks) {
    const int k0 = ks * 32;
    #pragma unroll
    for (int m = 0; m < 3; ++m) {
      const _Float16* ap = A + (long)(m * 16 + lr) * 64 + k0 + hi * 16;
      const v16h a = load_frag(ap, ap + 8);
      const _Float16* bp = &VsT[wv * 16 + lr][k0 + hi * 16];
      const v16h bfr = load_frag(bp, bp + 8);
      acc[m] = wmma_f16(a, bfr, acc[m]);
    }
  }
  #pragma unroll
  for (int m = 0; m < 3; ++m) {
    const long row = (long)b * CDIM + h * HD + m * 16 + hi * 8;
    const int  gc  = col0 + wv * 16 + lr;
    #pragma unroll
    for (int r = 0; r < 8; ++r)
      out[(row + r) * (long)HW + gc] = acc[m][r];
  }
}

// ---------------------------------------------------------------------------
// Host orchestration.  Workspace layout (bytes):
//   R1 qkv_x  f32 [4,1152,HW]   301,989,888   (reused as attn output later)
//   R2 qm     f32 [4, 384,HW]   100,663,296
//   R3 kv_dw  f32 [4, 768,HW]   201,326,592
//   R4 qm_dw  f32 [4, 384,HW]   100,663,296
//   R5 kH     f16 [4, 384,HW]    50,331,648
//   R6 qmH    f16 [4, 384,HW]    50,331,648
//   R7 vH     f16 [4, 384,HW]    50,331,648
//   R8 attnH  f16 [4,8,48,64]        196,608
// ---------------------------------------------------------------------------
extern "C" void kernel_launch(void* const* d_in, const int* in_sizes, int n_in,
                              void* d_out, int out_size, void* d_ws, size_t ws_size,
                              hipStream_t stream) {
  const float* x      = (const float*)d_in[0];
  const float* msg    = (const float*)d_in[1];
  const float* w_qkv  = (const float*)d_in[2];
  const float* w_dw   = (const float*)d_in[3];
  const float* w_proj = (const float*)d_in[4];
  const float* temp   = (const float*)d_in[5];
  float* out = (float*)d_out;

  char* ws = (char*)d_ws;
  float*     qkvX  = (float*)(ws);
  float*     qM    = (float*)(ws + 301989888L);
  float*     kvDW  = (float*)(ws + 301989888L + 100663296L);
  float*     qmDW  = (float*)(ws + 301989888L + 100663296L + 201326592L);
  _Float16*  kH    = (_Float16*)(ws + 301989888L + 100663296L + 201326592L + 100663296L);
  _Float16*  qmH   = (_Float16*)((char*)kH  + 50331648L);
  _Float16*  vH    = (_Float16*)((char*)qmH + 50331648L);
  _Float16*  attnH = (_Float16*)((char*)vH  + 50331648L);
  float*     aOut  = qkvX;   // alias: qkv_x dead after depthwise conv

  const long sX = (long)CDIM * HW;

  // 1) qkv of x : [1152x384] x [384xHW]
  gemm_wx_f32<<<dim3(HW / 128, 1152 / 64, NB), 256, 0, stream>>>(
      w_qkv, x, qkvX, 1152, CDIM, HW, sX, (long)1152 * HW);
  // 2) q of message (reference discards k,v of message)
  gemm_wx_f32<<<dim3(HW / 128, CDIM / 64, NB), 256, 0, stream>>>(
      w_qkv, msg, qM, CDIM, CDIM, HW, sX, sX);

  // 3) depthwise 3x3 on k,v channels of qkv_x
  {
    const long total = (long)NB * 768 * HW;
    dwconv3x3<<<(unsigned)(total / 256), 256, 0, stream>>>(qkvX, kvDW, w_dw, 384, 1152, 768);
  }
  // 4) depthwise 3x3 on q channels of message
  {
    const long total = (long)NB * CDIM * HW;
    dwconv3x3<<<(unsigned)(total / 256), 256, 0, stream>>>(qM, qmDW, w_dw, 0, CDIM, CDIM);
  }

  // 5) L2-normalize k and q_m rows; convert v; all to f16
  row_to_half<<<dim3(CDIM, NB), 256, 0, stream>>>(kvDW, kH, 768, 0, CDIM, 1);
  row_to_half<<<dim3(CDIM, NB), 256, 0, stream>>>(kvDW, vH, 768, 384, CDIM, 0);
  row_to_half<<<dim3(CDIM, NB), 256, 0, stream>>>(qmDW, qmH, CDIM, 0, CDIM, 1);

  // 6) attention scores + softmax
  attn_qk_softmax<<<dim3(NHEAD, NB), 256, 0, stream>>>(qmH, kH, temp, attnH);

  // 7) attn . V
  attn_v<<<dim3(HW / 128, NHEAD, NB), 256, 0, stream>>>(attnH, vH, aOut);

  // 8) output projection
  gemm_wx_f32<<<dim3(HW / 128, CDIM / 64, NB), 256, 0, stream>>>(
      w_proj, aOut, out, CDIM, CDIM, HW, sX, sX);
}
